// AttnBlock_57002805952558
// MI455X (gfx1250) — matrix operand order repaired and checked by reading, not compiled, not run on hardware
//
#include <hip/hip_runtime.h>
#include <hip/hip_bf16.h>

// ---------------------------------------------------------------------------
// AttnBlock for MI455X (gfx1250, wave32, WMMA 16x16x32 f16)
//   x[8,512,2048] -> GN -> q,k,v proj -> flash attention -> out proj + resid
// All GEMMs go through v_wmma_f32_16x16x32_f16.
// ---------------------------------------------------------------------------

typedef __attribute__((ext_vector_type(16))) _Float16 v16h;
typedef __attribute__((ext_vector_type(8)))  _Float16 v8h;
typedef __attribute__((ext_vector_type(8)))  float    v8f;

union V16 { v16h v; v8h h[2]; };

#define CB   8        // batch
#define CC   512      // channels
#define CL   2048     // sequence length
#define CNH  4        // heads
#define CHD  128      // head dim
#define CPG  128      // channels per group (512/4)

// ---- WMMA fragment loaders (ISA 7.12.2 layouts, wave32) --------------------

// A fragment: 16x32 f16, row-major source, ld in halfs.
// lane: M = m0 + (lane&15); element e: K = k0 + (e>>3)*16 + (lane>>4)*8 + (e&7)
__device__ __forceinline__ v16h load_a_frag(const _Float16* p, int ld, int m0, int k0) {
    int lane = threadIdx.x & 31;
    const _Float16* base = p + (size_t)(m0 + (lane & 15)) * ld + (k0 + ((lane >> 4) << 3));
    V16 u;
    u.h[0] = *(const v8h*)(base);
    u.h[1] = *(const v8h*)(base + 16);
    return u.v;
}

// B fragment: 32x16 f16 from "K-contiguous" storage: addr = col*ldc + k.
// lane: N = n0 + (lane&15); element e: K = k0 + (e>>3)*16 + (lane>>4)*8 + (e&7)
__device__ __forceinline__ v16h load_b_frag(const _Float16* p, int ldc, int n0, int k0) {
    int lane = threadIdx.x & 31;
    const _Float16* base = p + (size_t)(n0 + (lane & 15)) * ldc + (k0 + ((lane >> 4) << 3));
    V16 u;
    u.h[0] = *(const v8h*)(base);
    u.h[1] = *(const v8h*)(base + 16);
    return u.v;
}

#define WMMA_F16(a, b, c) \
    __builtin_amdgcn_wmma_f32_16x16x32_f16(false, (a), false, (b), (short)0, (c), false, false)

// ---- Kernel 1: GroupNorm, fp32 [B,C,L] -> f16 transposed h_t[B,L,C] --------
__global__ __launch_bounds__(256) void gn_kernel(const float* __restrict__ x,
                                                 const float* __restrict__ gsc,
                                                 const float* __restrict__ gbi,
                                                 _Float16* __restrict__ h_t) {
    int b = blockIdx.x >> 2;
    int g = blockIdx.x & 3;
    const float* xb = x + ((size_t)b * CC + (size_t)g * CPG) * CL;
    const int N = CPG * CL;
    int tid = threadIdx.x;

    float s = 0.f, ss = 0.f;
    for (int i = tid; i < N; i += 256) { float v = xb[i]; s += v; ss += v * v; }
    __shared__ float rs[256], rss[256];
    rs[tid] = s; rss[tid] = ss;
    __syncthreads();
    for (int off = 128; off > 0; off >>= 1) {
        if (tid < off) { rs[tid] += rs[tid + off]; rss[tid] += rss[tid + off]; }
        __syncthreads();
    }
    float mean = rs[0] / (float)N;
    float var  = rss[0] / (float)N - mean * mean;
    float rstd = rsqrtf(var + 1e-6f);

    for (int i = tid; i < N; i += 256) {
        int c = g * CPG + (i >> 11);      // i / 2048
        int l = i & (CL - 1);
        float v = (xb[i] - mean) * rstd * gsc[c] + gbi[c];
        h_t[((size_t)b * CL + l) * CC + c] = (_Float16)v;
    }
}

// ---- Kernel 2: convert all 4 weight matrices fp32 -> f16 -------------------
__global__ __launch_bounds__(256) void cvt_kernel(const float* __restrict__ a,
                                                  const float* __restrict__ b,
                                                  const float* __restrict__ c,
                                                  const float* __restrict__ d,
                                                  _Float16* __restrict__ o) {
    int i = blockIdx.x * 256 + threadIdx.x;
    const int NW = CC * CC;
    if (i < NW) {
        o[i]          = (_Float16)a[i];
        o[i + NW]     = (_Float16)b[i];
        o[i + 2 * NW] = (_Float16)c[i];
        o[i + 3 * NW] = (_Float16)d[i];
    }
}

// ---- Kernel 3: GEMM  O[m,n] = sum_k W[m,k] * H[b][n*K+k] + bias[m] ---------
// M=512, N=2048, K=512 per batch. Block = 8 waves -> 64(M) x 128(N) tile.
// mode 0: f16 out transposed [B,N,M], scaled   (q_t / k_t)
// mode 1: f16 out [B,M,N]                      (v)
// mode 2: f32 out [B,M,N] + residual x         (final output)
__global__ __launch_bounds__(256) void gemm_kernel(const _Float16* __restrict__ W,
                                                   const _Float16* __restrict__ H,
                                                   const float* __restrict__ bias,
                                                   void* __restrict__ outp,
                                                   const float* __restrict__ resid,
                                                   float scale, int mode) {
    int b = blockIdx.z;
    int n_blk = blockIdx.x * 128;
    int m_blk = blockIdx.y * 64;
    int tid = threadIdx.x, lane = tid & 31, w = tid >> 5;
    int wr = w >> 2, wc = w & 3;                 // 2 x 4 wave grid
    int m_w = m_blk + wr * 32;
    int n_w = n_blk + wc * 32;
    const _Float16* Hb = H + (size_t)b * CL * CC;

    v8f acc00 = {}, acc01 = {}, acc10 = {}, acc11 = {};
    for (int k = 0; k < CC; k += 32) {
        v16h a0 = load_a_frag(W, CC, m_w,      k);
        v16h a1 = load_a_frag(W, CC, m_w + 16, k);
        v16h b0 = load_b_frag(Hb, CC, n_w,      k);
        v16h b1 = load_b_frag(Hb, CC, n_w + 16, k);
        acc00 = WMMA_F16(a0, b0, acc00);
        acc01 = WMMA_F16(a0, b1, acc01);
        acc10 = WMMA_F16(a1, b0, acc10);
        acc11 = WMMA_F16(a1, b1, acc11);
    }

    v8f* accs[4] = { &acc00, &acc01, &acc10, &acc11 };
    #pragma unroll
    for (int t = 0; t < 4; t++) {
        int i = t >> 1, j = t & 1;
        v8f av = *accs[t];
        int Mb = m_w + 16 * i + ((lane >> 4) << 3);
        int Nc = n_w + 16 * j + (lane & 15);
        if (mode == 0) {
            _Float16* o = (_Float16*)outp;
            v8h st;
            #pragma unroll
            for (int v = 0; v < 8; v++)
                st[v] = (_Float16)((av[v] + bias[Mb + v]) * scale);
            *(v8h*)(o + ((size_t)b * CL + Nc) * CC + Mb) = st;
        } else if (mode == 1) {
            _Float16* o = (_Float16*)outp;
            #pragma unroll
            for (int v = 0; v < 8; v++)
                o[((size_t)b * CC + Mb + v) * CL + Nc] = (_Float16)(av[v] + bias[Mb + v]);
        } else {
            float* o = (float*)outp;
            #pragma unroll
            for (int v = 0; v < 8; v++) {
                size_t idx = ((size_t)b * CC + Mb + v) * CL + Nc;
                o[idx] = av[v] + bias[Mb + v] + resid[idx];
            }
        }
    }
}

// ---- Kernel 4: flash attention -------------------------------------------
// q_t,k_t: f16 [B,L,C] (scale already folded into q). v: f16 [B,C,L].
// Block handles 64 queries of one (b,h). 8 waves = 4 query-rows x 2 d-halves.
// Streams 32-key tiles: S = Q^T K (4 wmma), online softmax in LDS, O += P*V (4 wmma).
__global__ __launch_bounds__(256) void attn_kernel(const _Float16* __restrict__ q_t,
                                                   const _Float16* __restrict__ k_t,
                                                   const _Float16* __restrict__ vm,
                                                   _Float16* __restrict__ attn_t) {
    int b = blockIdx.z, h = blockIdx.y, q0 = blockIdx.x * 64;
    int tid = threadIdx.x, lane = tid & 31, w = tid >> 5;
    int wr = w >> 1, wc = w & 1;

    __shared__ __align__(16) float    S[64][32];
    __shared__ __align__(16) _Float16 P[64][32];
    __shared__ float mrow[64], lrow[64], fac[64];

    const _Float16* qb = q_t + (size_t)b * CL * CC;
    const _Float16* kp = k_t + (size_t)b * CL * CC;
    const _Float16* vb = vm  + (size_t)b * CC * CL;

    v16h qf[4];
    #pragma unroll
    for (int kk = 0; kk < 4; kk++)
        qf[kk] = load_a_frag(qb, CC, q0 + wr * 16, h * CHD + kk * 32);

    v8f acc0 = {}, acc1 = {}, acc2 = {}, acc3 = {};
    if (tid < 64) { mrow[tid] = -1e30f; lrow[tid] = 0.f; }
    __syncthreads();

    int rbase = wr * 16 + ((lane >> 4) << 3);
    int cidx  = wc * 16 + (lane & 15);

    for (int k0 = 0; k0 < CL; k0 += 32) {
        // ---- scores: wave computes S[16 rows, 16 cols] over d=128
        v8f s = {};
        #pragma unroll
        for (int kk = 0; kk < 4; kk++) {
            v16h bf = load_b_frag(kp, CC, k0 + wc * 16, h * CHD + kk * 32);
            s = WMMA_F16(qf[kk], bf, s);
        }
        #pragma unroll
        for (int v = 0; v < 8; v++) S[rbase + v][cidx] = s[v];
        __syncthreads();

        // ---- online softmax stats + P (threads 0..63, one row each)
        if (tid < 64) {
            float mo = mrow[tid], mx = mo;
            for (int j = 0; j < 32; j++) mx = fmaxf(mx, S[tid][j]);
            float corr = __expf(mo - mx);
            float ls = lrow[tid] * corr;
            for (int j = 0; j < 32; j++) {
                float p = __expf(S[tid][j] - mx);
                ls += p;
                P[tid][j] = (_Float16)p;
            }
            mrow[tid] = mx; lrow[tid] = ls; fac[tid] = corr;
        }
        __syncthreads();

        // ---- rescale accumulators and O += P * V
        float f[8];
        #pragma unroll
        for (int v = 0; v < 8; v++) f[v] = fac[rbase + v];
        #pragma unroll
        for (int v = 0; v < 8; v++) {
            acc0[v] *= f[v]; acc1[v] *= f[v]; acc2[v] *= f[v]; acc3[v] *= f[v];
        }
        v16h pf = load_a_frag(&P[0][0], 32, wr * 16, 0);
        {
            v16h vf0 = load_b_frag(vb, CL, h * CHD + wc * 64 +  0, k0);
            acc0 = WMMA_F16(pf, vf0, acc0);
            v16h vf1 = load_b_frag(vb, CL, h * CHD + wc * 64 + 16, k0);
            acc1 = WMMA_F16(pf, vf1, acc1);
            v16h vf2 = load_b_frag(vb, CL, h * CHD + wc * 64 + 32, k0);
            acc2 = WMMA_F16(pf, vf2, acc2);
            v16h vf3 = load_b_frag(vb, CL, h * CHD + wc * 64 + 48, k0);
            acc3 = WMMA_F16(pf, vf3, acc3);
        }
        // next-iteration __syncthreads (after S write) protects P/fac reuse
    }

    float rl[8];
    #pragma unroll
    for (int v = 0; v < 8; v++) rl[v] = 1.0f / lrow[rbase + v];

    v8f* accs[4] = { &acc0, &acc1, &acc2, &acc3 };
    #pragma unroll
    for (int t = 0; t < 4; t++) {
        int d = h * CHD + wc * 64 + t * 16 + (lane & 15);
        v8f av = *accs[t];
        #pragma unroll
        for (int v = 0; v < 8; v++)
            attn_t[((size_t)b * CL + q0 + rbase + v) * CC + d] = (_Float16)(av[v] * rl[v]);
    }
}

// ---------------------------------------------------------------------------
extern "C" void kernel_launch(void* const* d_in, const int* in_sizes, int n_in,
                              void* d_out, int out_size, void* d_ws, size_t ws_size,
                              hipStream_t stream) {
    (void)in_sizes; (void)n_in; (void)out_size; (void)ws_size;
    const float* x   = (const float*)d_in[0];
    const float* gsc = (const float*)d_in[1];
    const float* gbi = (const float*)d_in[2];
    const float* wq  = (const float*)d_in[3];
    const float* bq  = (const float*)d_in[4];
    const float* wk  = (const float*)d_in[5];
    const float* bk  = (const float*)d_in[6];
    const float* wv  = (const float*)d_in[7];
    const float* bv  = (const float*)d_in[8];
    const float* wo  = (const float*)d_in[9];
    const float* bo  = (const float*)d_in[10];

    char* ws = (char*)d_ws;
    const size_t MB = 1024 * 1024;
    _Float16* h_t  = (_Float16*)(ws);             // [B,L,C] f16  16 MB
    _Float16* q_t  = (_Float16*)(ws + 16 * MB);   // [B,L,C] f16  16 MB
    _Float16* k_t  = (_Float16*)(ws + 32 * MB);   // [B,L,C] f16  16 MB
    _Float16* v_m  = (_Float16*)(ws + 48 * MB);   // [B,C,L] f16  16 MB
    _Float16* a_t  = (_Float16*)(ws + 64 * MB);   // [B,L,C] f16  16 MB
    _Float16* w16  = (_Float16*)(ws + 80 * MB);   // 4 x 512x512 f16  2 MB
    const int NW = CC * CC;

    gn_kernel<<<CB * 4, 256, 0, stream>>>(x, gsc, gbi, h_t);
    cvt_kernel<<<(NW + 255) / 256, 256, 0, stream>>>(wq, wk, wv, wo, w16);

    dim3 gg(CL / 128, CC / 64, CB);
    const float qscale = 0.08838834764831845f;   // 1/sqrt(128), folded into q
    gemm_kernel<<<gg, 256, 0, stream>>>(w16,          h_t, bq, (void*)q_t, nullptr, qscale, 0);
    gemm_kernel<<<gg, 256, 0, stream>>>(w16 + NW,     h_t, bk, (void*)k_t, nullptr, 1.0f,   0);
    gemm_kernel<<<gg, 256, 0, stream>>>(w16 + 2 * NW, h_t, bv, (void*)v_m, nullptr, 1.0f,   1);

    attn_kernel<<<dim3(CL / 64, CNH, CB), 256, 0, stream>>>(q_t, k_t, v_m, a_t);

    gemm_kernel<<<gg, 256, 0, stream>>>(w16 + 3 * NW, a_t, bo, d_out, x, 1.0f, 2);
}
